// MultiHeadAttention_25718264169260
// MI455X (gfx1250) — matrix operand order repaired and checked by reading, not compiled, not run on hardware
//
#include <hip/hip_runtime.h>
#include <hip/hip_bf16.h>

#define EDIM 1024
#define SLEN 2048
#define BATCH 2
#define NH 16
#define HD 64
#define MTOT (BATCH * SLEN)   // 4096

typedef __attribute__((ext_vector_type(16))) _Float16 v16h;
typedef __attribute__((ext_vector_type(8)))  _Float16 v8h;
typedef __attribute__((ext_vector_type(4)))  _Float16 v4h;
typedef __attribute__((ext_vector_type(8)))  float    v8f;
typedef __attribute__((ext_vector_type(4)))  float    v4f;

union HFrag { v16h v; v8h h8[2]; v4h h4[4]; };

// A-matrix fragment (16x32 f16, M=lane&15): lanes 0-15 hold K {0..7, 16..23},
// lanes 16-31 hold K {8..15, 24..31}  (ISA 7.12.2, 16-bit A table)
__device__ __forceinline__ v16h frag_a_f16(const _Float16* base, int ld, int lane) {
  const _Float16* p = base + (size_t)(lane & 15) * ld + ((lane >> 4) & 1) * 8;
  HFrag f;
  f.h8[0] = *(const v8h*)(p);
  f.h8[1] = *(const v8h*)(p + 16);
  return f.v;
}
// B-matrix fragment (32x16 f16, N=lane&15): lanes 0-15 hold K {0..7, 16..23},
// lanes 16-31 hold K {8..15, 24..31}.  `base` = element (k=0,n=0) of [N,K] row-major.
__device__ __forceinline__ v16h frag_b_f16(const _Float16* base, int ld, int lane) {
  const _Float16* p = base + (size_t)(lane & 15) * ld + ((lane >> 4) & 1) * 8;
  HFrag f;
  f.h8[0] = *(const v8h*)(p);
  f.h8[1] = *(const v8h*)(p + 16);
  return f.v;
}
__device__ __forceinline__ v8f wmma_f16(v16h a, v16h b, v8f c) {
  return __builtin_amdgcn_wmma_f32_16x16x32_f16(false, a, false, b, (short)0, c,
                                                false, false);
}

// 32x64 wave-tile GEMM inner loop, k-unrolled by 2 with two explicit buffer
// sets so the software pipeline needs no register rotation movs.
// A rows at Ar0/Ar1 (16 rows each, ld=EDIM); B columns at Wn (4 groups of 16
// rows of a [N,K] row-major matrix).
__device__ __forceinline__ void gemm_32x64_f16(
    const _Float16* __restrict__ Ar0, const _Float16* __restrict__ Ar1,
    const _Float16* __restrict__ Wn, int lane, v8f acc0[4], v8f acc1[4]) {
  v16h a0A = frag_a_f16(Ar0, EDIM, lane);
  v16h a1A = frag_a_f16(Ar1, EDIM, lane);
  v16h bA[4], bB[4];
#pragma unroll
  for (int j = 0; j < 4; ++j)
    bA[j] = frag_b_f16(Wn + (size_t)16 * j * EDIM, EDIM, lane);
  v16h a0B = frag_a_f16(Ar0 + 32, EDIM, lane);
  v16h a1B = frag_a_f16(Ar1 + 32, EDIM, lane);
#pragma unroll
  for (int j = 0; j < 4; ++j)
    bB[j] = frag_b_f16(Wn + (size_t)16 * j * EDIM + 32, EDIM, lane);

  for (int k0 = 64; k0 < EDIM; k0 += 64) {
    // stage A: consume buffer A (data k0-64), refill with k0
#pragma unroll
    for (int j = 0; j < 4; ++j) {
      acc0[j] = wmma_f16(a0A, bA[j], acc0[j]);
      acc1[j] = wmma_f16(a1A, bA[j], acc1[j]);
    }
    a0A = frag_a_f16(Ar0 + k0, EDIM, lane);
    a1A = frag_a_f16(Ar1 + k0, EDIM, lane);
#pragma unroll
    for (int j = 0; j < 4; ++j)
      bA[j] = frag_b_f16(Wn + (size_t)16 * j * EDIM + k0, EDIM, lane);
    // stage B: consume buffer B (data k0-32), refill with k0+32
#pragma unroll
    for (int j = 0; j < 4; ++j) {
      acc0[j] = wmma_f16(a0B, bB[j], acc0[j]);
      acc1[j] = wmma_f16(a1B, bB[j], acc1[j]);
    }
    a0B = frag_a_f16(Ar0 + k0 + 32, EDIM, lane);
    a1B = frag_a_f16(Ar1 + k0 + 32, EDIM, lane);
#pragma unroll
    for (int j = 0; j < 4; ++j)
      bB[j] = frag_b_f16(Wn + (size_t)16 * j * EDIM + k0 + 32, EDIM, lane);
  }
#pragma unroll
  for (int j = 0; j < 4; ++j) {
    acc0[j] = wmma_f16(a0A, bA[j], acc0[j]);
    acc1[j] = wmma_f16(a1A, bA[j], acc1[j]);
  }
#pragma unroll
  for (int j = 0; j < 4; ++j) {
    acc0[j] = wmma_f16(a0B, bB[j], acc0[j]);
    acc1[j] = wmma_f16(a1B, bB[j], acc1[j]);
  }
}

// ---------------------------------------------------------------------------
// Kernel 0: one-pass f32 -> f16 conversion (streamed, 8 elems/thread).
// ---------------------------------------------------------------------------
__global__ __launch_bounds__(256) void cvt_f16_kernel(
    const float* __restrict__ src, _Float16* __restrict__ dst, int n8) {
  int i = blockIdx.x * blockDim.x + threadIdx.x;
  if (i < n8) {
    const float* s = src + (size_t)i * 8;
    union { v8h v; v4h h[2]; } u;
    u.h[0] = __builtin_convertvector(*(const v4f*)(s),     v4h);
    u.h[1] = __builtin_convertvector(*(const v4f*)(s + 4), v4h);
    *(v8h*)(dst + (size_t)i * 8) = u.v;
  }
}

// ---------------------------------------------------------------------------
// Kernel 1: Q/K/V projections, pure-f16 WMMA GEMM.  C[m,n]=sum_k X[m,k]W[n,k]
// Q stored [B,H,S,D] scaled 1/sqrt(D); K stored [B,H,S,D]; V stored [B,H,D,S].
// ---------------------------------------------------------------------------
__global__ __launch_bounds__(256) void qkv_proj_kernel(
    const _Float16* __restrict__ Xq, const _Float16* __restrict__ Xk,
    const _Float16* __restrict__ Xv, const _Float16* __restrict__ Wq,
    const _Float16* __restrict__ Wk, const _Float16* __restrict__ Wv,
    _Float16* __restrict__ Oq, _Float16* __restrict__ Ok,
    _Float16* __restrict__ Ovt) {
  const int lane = threadIdx.x & 31;
  const int wave = threadIdx.x >> 5;
  const int m0 = blockIdx.x * 32;
  const int n0 = blockIdx.y * 512 + wave * 64;
  const int mode = blockIdx.z;  // 0=Q, 1=K, 2=V
  const _Float16* X = (mode == 0) ? Xq : (mode == 1 ? Xk : Xv);
  const _Float16* W = (mode == 0) ? Wq : (mode == 1 ? Wk : Wv);

  v8f acc0[4] = {}, acc1[4] = {};
  gemm_32x64_f16(X + (size_t)m0 * EDIM, X + (size_t)(m0 + 16) * EDIM,
                 W + (size_t)n0 * EDIM, lane, acc0, acc1);

  const float scale = (mode == 0) ? 0.125f : 1.0f;  // fold 1/sqrt(64) into Q
  const int hi = (lane >> 4) & 1;
#pragma unroll
  for (int half = 0; half < 2; ++half) {
    const v8f* acc = half ? acc1 : acc0;
#pragma unroll
    for (int j = 0; j < 4; ++j) {
#pragma unroll
      for (int r = 0; r < 8; ++r) {
        int m = m0 + half * 16 + r + hi * 8;   // token index
        int n = n0 + 16 * j + (lane & 15);     // embed index
        int b = m >> 11, s = m & (SLEN - 1);
        int h = n >> 6,  d = n & (HD - 1);
        _Float16 val = (_Float16)(acc[j][r] * scale);
        if (mode == 2)
          Ovt[((size_t)(b * NH + h) * HD + d) * SLEN + s] = val;
        else if (mode == 1)
          Ok[((size_t)(b * NH + h) * SLEN + s) * HD + d] = val;
        else
          Oq[((size_t)(b * NH + h) * SLEN + s) * HD + d] = val;
      }
    }
  }
}

// ---------------------------------------------------------------------------
// Kernel 2: flash attention. One wave owns 16 query rows of one (b,h);
// per 32 keys: 4 score WMMAs, online softmax (shfl_xor row reductions),
// LDS round-trip to relayout P (D-layout -> A-layout), 4 PV WMMAs.
// ---------------------------------------------------------------------------
__global__ __launch_bounds__(128) void flash_attn_kernel(
    const _Float16* __restrict__ Qg, const _Float16* __restrict__ Kg,
    const _Float16* __restrict__ Vtg, _Float16* __restrict__ Ctx) {
  __shared__ __align__(16) _Float16 plds[4][16][40];  // +8 halfs row pad
  const int lane = threadIdx.x & 31;
  const int wave = threadIdx.x >> 5;
  const int bh = blockIdx.y;
  const int b = bh >> 4, h = bh & 15;
  const int qt = blockIdx.x * 4 + wave;
  const int hi = (lane >> 4) & 1;

  const _Float16* Qh  = Qg  + (size_t)bh * SLEN * HD;
  const _Float16* Kh  = Kg  + (size_t)bh * SLEN * HD;
  const _Float16* Vth = Vtg + (size_t)bh * HD * SLEN;

  v16h qa0 = frag_a_f16(Qh + (size_t)qt * 16 * HD, HD, lane);       // K 0..31
  v16h qa1 = frag_a_f16(Qh + (size_t)qt * 16 * HD + 32, HD, lane);  // K 32..63

  v8f o[4] = {};
  float mrun[8], lrun[8];
#pragma unroll
  for (int r = 0; r < 8; ++r) { mrun[r] = -1e30f; lrun[r] = 0.0f; }

  for (int kb = 0; kb < SLEN; kb += 32) {
    // prefetch next 32-key block of K (32 rows x 128B) and V^T (64 rows x 64B)
    if (kb + 32 < SLEN) {
      __builtin_prefetch(Kh + (size_t)(kb + 32) * HD + lane * 64, 0, 1);
      __builtin_prefetch(Vth + (size_t)lane * SLEN + kb + 32, 0, 1);
      __builtin_prefetch(Vth + (size_t)(lane + 32) * SLEN + kb + 32, 0, 1);
    }
    v16h k00 = frag_b_f16(Kh + (size_t)kb * HD,             HD, lane);
    v16h k01 = frag_b_f16(Kh + (size_t)kb * HD + 32,        HD, lane);
    v16h k10 = frag_b_f16(Kh + (size_t)(kb + 16) * HD,      HD, lane);
    v16h k11 = frag_b_f16(Kh + (size_t)(kb + 16) * HD + 32, HD, lane);
    v8f sA = {}, sB = {};
    sA = wmma_f16(qa0, k00, sA);
    sA = wmma_f16(qa1, k01, sA);
    sB = wmma_f16(qa0, k10, sB);
    sB = wmma_f16(qa1, k11, sB);

    float corr[8];
#pragma unroll
    for (int r = 0; r < 8; ++r) {
      float t = fmaxf(sA[r], sB[r]);
      t = fmaxf(t, __shfl_xor(t, 1, 32));
      t = fmaxf(t, __shfl_xor(t, 2, 32));
      t = fmaxf(t, __shfl_xor(t, 4, 32));
      t = fmaxf(t, __shfl_xor(t, 8, 32));
      float mn = fmaxf(mrun[r], t);
      corr[r] = __expf(mrun[r] - mn);
      mrun[r] = mn;
      float pa = __expf(sA[r] - mn);
      float pb = __expf(sB[r] - mn);
      plds[wave][r + hi * 8][(lane & 15)]      = (_Float16)pa;
      plds[wave][r + hi * 8][(lane & 15) + 16] = (_Float16)pb;
      float rs = pa + pb;
      rs += __shfl_xor(rs, 1, 32);
      rs += __shfl_xor(rs, 2, 32);
      rs += __shfl_xor(rs, 4, 32);
      rs += __shfl_xor(rs, 8, 32);
      lrun[r] = lrun[r] * corr[r] + rs;
    }
#pragma unroll
    for (int j = 0; j < 4; ++j)
#pragma unroll
      for (int r = 0; r < 8; ++r) o[j][r] *= corr[r];

    // reload P in A-fragment layout (same-wave LDS is in-order; compiler
    // inserts s_wait_dscnt before first use)
    const _Float16* pr = &plds[wave][lane & 15][hi * 8];
    HFrag pf;
    pf.h8[0] = *(const v8h*)(pr);
    pf.h8[1] = *(const v8h*)(pr + 16);

#pragma unroll
    for (int j = 0; j < 4; ++j) {
      v16h vf = frag_b_f16(Vth + (size_t)(16 * j) * SLEN + kb, SLEN, lane);
      o[j] = wmma_f16(pf.v, vf, o[j]);
    }
  }

  float inv[8];
#pragma unroll
  for (int r = 0; r < 8; ++r) inv[r] = 1.0f / lrun[r];
#pragma unroll
  for (int j = 0; j < 4; ++j) {
#pragma unroll
    for (int r = 0; r < 8; ++r) {
      int srow = qt * 16 + r + hi * 8;
      int col = h * HD + 16 * j + (lane & 15);
      Ctx[(size_t)(b * SLEN + srow) * EDIM + col] = (_Float16)(o[j][r] * inv[r]);
    }
  }
}

// ---------------------------------------------------------------------------
// Kernel 3: output projection.  out[m,n] = sum_k ctx[m,k]*Wo[n,k] + bo[n]
// ---------------------------------------------------------------------------
__global__ __launch_bounds__(256) void out_proj_kernel(
    const _Float16* __restrict__ Ctx, const _Float16* __restrict__ Wo16,
    const float* __restrict__ bo, float* __restrict__ Out) {
  const int lane = threadIdx.x & 31;
  const int wave = threadIdx.x >> 5;
  const int m0 = blockIdx.x * 32;
  const int n0 = blockIdx.y * 512 + wave * 64;

  v8f acc0[4] = {}, acc1[4] = {};
  gemm_32x64_f16(Ctx + (size_t)m0 * EDIM, Ctx + (size_t)(m0 + 16) * EDIM,
                 Wo16 + (size_t)n0 * EDIM, lane, acc0, acc1);

  const int hi = (lane >> 4) & 1;
#pragma unroll
  for (int half = 0; half < 2; ++half) {
    const v8f* acc = half ? acc1 : acc0;
#pragma unroll
    for (int j = 0; j < 4; ++j) {
      int n = n0 + 16 * j + (lane & 15);
      float bias = bo[n];
#pragma unroll
      for (int r = 0; r < 8; ++r) {
        int m = m0 + half * 16 + r + hi * 8;
        Out[(size_t)m * EDIM + n] = acc[j][r] + bias;
      }
    }
  }
}

extern "C" void kernel_launch(void* const* d_in, const int* in_sizes, int n_in,
                              void* d_out, int out_size, void* d_ws,
                              size_t ws_size, hipStream_t stream) {
  (void)in_sizes; (void)n_in; (void)out_size; (void)ws_size;
  const float* key   = (const float*)d_in[0];
  const float* query = (const float*)d_in[1];
  const float* value = (const float*)d_in[2];
  const float* Wq    = (const float*)d_in[3];
  const float* Wk    = (const float*)d_in[4];
  const float* Wv    = (const float*)d_in[5];
  const float* Wo    = (const float*)d_in[6];
  const float* bo    = (const float*)d_in[7];
  float* out = (float*)d_out;

  // f16 workspace layout (halfs):
  // xq|xk|xv [M,E] (3 x 4Mi) | wq|wk|wv|wo [E,E] (4 x 1Mi)
  // | q|k [B,H,S,D], v^T [B,H,D,S] (3 x 4Mi) | ctx [B,S,E] (4Mi)  = 64 MB
  _Float16* w = (_Float16*)d_ws;
  const size_t XS = (size_t)MTOT * EDIM;    // 4 Mi halfs
  const size_t WSZ = (size_t)EDIM * EDIM;   // 1 Mi halfs
  _Float16* xq16 = w;
  _Float16* xk16 = xq16 + XS;
  _Float16* xv16 = xk16 + XS;
  _Float16* wq16 = xv16 + XS;
  _Float16* wk16 = wq16 + WSZ;
  _Float16* wv16 = wk16 + WSZ;
  _Float16* wo16 = wv16 + WSZ;
  _Float16* q    = wo16 + WSZ;
  _Float16* k    = q + XS;
  _Float16* vt   = k + XS;
  _Float16* ctx  = vt + XS;

  auto cvt = [&](const float* s, _Float16* d, size_t n) {
    int n8 = (int)(n / 8);
    cvt_f16_kernel<<<(n8 + 255) / 256, 256, 0, stream>>>(s, d, n8);
  };
  cvt(query, xq16, XS);
  cvt(key,   xk16, XS);
  cvt(value, xv16, XS);
  cvt(Wq, wq16, WSZ);
  cvt(Wk, wk16, WSZ);
  cvt(Wv, wv16, WSZ);
  cvt(Wo, wo16, WSZ);

  dim3 gproj(MTOT / 32, EDIM / 512, 3);
  qkv_proj_kernel<<<gproj, 256, 0, stream>>>(xq16, xk16, xv16, wq16, wk16,
                                             wv16, q, k, vt);
  dim3 gattn(SLEN / 16 / 4, BATCH * NH);
  flash_attn_kernel<<<gattn, 128, 0, stream>>>(q, k, vt, ctx);
  dim3 gout(MTOT / 32, EDIM / 512);
  out_proj_kernel<<<gout, 256, 0, stream>>>(ctx, wo16, bo, out);
}
